// RnnLm_39264591020747
// MI455X (gfx1250) — compile-verified, matrix operand order checked by reading, not run on hardware
//
#include <hip/hip_runtime.h>

typedef _Float16 v8h  __attribute__((ext_vector_type(8)));
typedef _Float16 v16h __attribute__((ext_vector_type(16)));
typedef float    v8f  __attribute__((ext_vector_type(8)));

#define VOCAB 10000
#define XDIM  128
#define HID   256
#define BATCH 128
#define SEQ   50

// ---------------- prep kernels ----------------

// W_ihT[x][h] = W_ih[h][x]   (128x256 f32, for coalesced column access)
__global__ void k_transpose_wih(const float* __restrict__ W_ih,
                                float* __restrict__ W_ihT) {
    int i = blockIdx.x * 256 + threadIdx.x;           // 0 .. 32767
    int h = i / XDIM, x = i % XDIM;
    W_ihT[x * HID + h] = W_ih[i];
}

// W_hhT[k][h] = W_hh[h][k]   (256x256 f32)
__global__ void k_transpose_whh(const float* __restrict__ W_hh,
                                float* __restrict__ W_hhT) {
    int i = blockIdx.x * 256 + threadIdx.x;           // 0 .. 65535
    int h = i / HID, k = i % HID;
    W_hhT[k * HID + h] = W_hh[i];
}

// W_aff f32 -> f16, row-major [V][H] unchanged (B fragment wants K-contiguous rows)
__global__ void k_convert_waff(const float* __restrict__ W_aff,
                               _Float16* __restrict__ wv16) {
    int i = blockIdx.x * 256 + threadIdx.x;           // 0 .. 2,559,999
    wv16[i] = (_Float16)W_aff[i];
}

// ---------------- recurrence: one block per batch row ----------------
__global__ __launch_bounds__(HID) void k_rnn_recur(
    const int* __restrict__ x, const float* __restrict__ emb,
    const float* __restrict__ W_ihT, const float* __restrict__ W_hhT,
    const float* __restrict__ b_ih, const float* __restrict__ b_hh,
    _Float16* __restrict__ hs16)
{
    __shared__ float h_lds[HID];
    __shared__ float e_lds[XDIM];
    const int j = threadIdx.x;
    const int b = blockIdx.x;
    const float bias = b_ih[j] + b_hh[j];
    h_lds[j] = 0.0f;
    __syncthreads();

    for (int s = 0; s < SEQ; ++s) {
        const int tok = x[b * SEQ + s];
        if (j < XDIM) e_lds[j] = emb[tok * XDIM + j];
        __syncthreads();

        float acc = bias;
        #pragma unroll 8
        for (int xk = 0; xk < XDIM; ++xk)
            acc = fmaf(e_lds[xk], W_ihT[xk * HID + j], acc);
        #pragma unroll 8
        for (int k = 0; k < HID; ++k)
            acc = fmaf(h_lds[k], W_hhT[k * HID + j], acc);

        const float hn = tanhf(acc);
        __syncthreads();                  // everyone done reading old h
        h_lds[j] = hn;
        hs16[(size_t)(b * SEQ + s) * HID + j] = (_Float16)hn;
        __syncthreads();                  // new h visible before next step
    }
}

// ---------------- output projection GEMM (WMMA f16 -> f32) ----------------
// out[6400,10000] = hs16[6400,256] @ wv16[10000,256]^T + b_aff
// Block tile: 64(M) x 80(N), 5 waves; wave tile 64(M) x 16(N).
// A strip (64x256 f16 = 32 KB) staged to LDS via global_load_async_to_lds_b128;
// A and B fragments both explicitly double-buffered so chunk k+1 loads retire
// under chunk k's WMMAs.
__global__ __launch_bounds__(160) void k_out_gemm(
    const _Float16* __restrict__ hs16, const _Float16* __restrict__ wv16,
    const float* __restrict__ b_aff, float* __restrict__ out)
{
    __shared__ _Float16 lds_a[64 * HID];              // 32 KB

    const int wv     = threadIdx.x >> 5;              // 0..4
    const int lane   = threadIdx.x & 31;
    const int mstrip = blockIdx.x / 125;              // 0..99
    const int nblk   = blockIdx.x % 125;              // 0..124
    const int m0 = mstrip * 64;
    const int n0 = (nblk * 5 + wv) * 16;
    const int r16 = lane & 15;                        // row within 16-tile
    const int hi8 = (lane >> 4) << 3;                 // 0 or 8

    // ---- async stage A strip: 32 KB = 128 threads x 16 B x 16 insts ----
    // IOFFSET is added to BOTH the LDS and global address (ISA 15.18.3).
    {
        const int t = threadIdx.x;
        if (t < 128) {
            unsigned long long gaddr =
                (unsigned long long)(hs16 + (size_t)m0 * HID) + (unsigned)t * 16u;
            unsigned int laddr = (unsigned int)(uintptr_t)lds_a + (unsigned)t * 16u;
            #pragma unroll
            for (int i = 0; i < 16; ++i) {
                asm volatile(
                    "global_load_async_to_lds_b128 %0, %1, off offset:%2"
                    :: "v"(laddr), "v"(gaddr), "n"(i * 2048) : "memory");
            }
        }
#if __has_builtin(__builtin_amdgcn_s_wait_asynccnt)
        __builtin_amdgcn_s_wait_asynccnt(0);
#else
        asm volatile("s_wait_asynccnt 0x0" ::: "memory");
#endif
        __syncthreads();
    }

    v8f acc[4] = {v8f{}, v8f{}, v8f{}, v8f{}};

    // B fragment: 32(K) x 16(N) tile of W_aff^T == 16 rows of W_aff
    auto loadB = [&](int kc) -> v16h {
        const _Float16* bp = wv16 + (size_t)(n0 + r16) * HID + kc * 32 + hi8;
        v8h bl = *(const v8h*)bp;
        v8h bh = *(const v8h*)(bp + 16);
        return __builtin_shufflevector(bl, bh,
            0,1,2,3,4,5,6,7,8,9,10,11,12,13,14,15);
    };
    // A fragment from LDS (ds_load_b128 x2)
    auto loadA = [&](int kc, int mt) -> v16h {
        const _Float16* ap = &lds_a[(mt * 16 + r16) * HID + kc * 32 + hi8];
        v8h al = *(const v8h*)ap;
        v8h ah = *(const v8h*)(ap + 16);
        return __builtin_shufflevector(al, ah,
            0,1,2,3,4,5,6,7,8,9,10,11,12,13,14,15);
    };

    v16h bf_cur = loadB(0);
    v16h af_cur[4], af_nxt[4];
    v16h bf_nxt;
    #pragma unroll
    for (int mt = 0; mt < 4; ++mt) af_cur[mt] = loadA(0, mt);

    #pragma unroll
    for (int kc = 0; kc < 8; ++kc) {
        if (kc < 7) {
            bf_nxt = loadB(kc + 1);                   // global, long latency first
            #pragma unroll
            for (int mt = 0; mt < 4; ++mt) af_nxt[mt] = loadA(kc + 1, mt);
        }
        #pragma unroll
        for (int mt = 0; mt < 4; ++mt) {
            acc[mt] = __builtin_amdgcn_wmma_f32_16x16x32_f16(
                /*neg_a=*/false, af_cur[mt], /*neg_b=*/false, bf_cur,
                /*c_mod=*/(short)0, acc[mt],
                /*reuse_a=*/false, /*reuse_b=*/false);
        }
        if (kc < 7) {
            bf_cur = bf_nxt;                          // register renames after unroll
            #pragma unroll
            for (int mt = 0; mt < 4; ++mt) af_cur[mt] = af_nxt[mt];
        }
    }

    const float bias = b_aff[n0 + r16];
    #pragma unroll
    for (int mt = 0; mt < 4; ++mt) {
        #pragma unroll
        for (int r = 0; r < 8; ++r) {
            const int m = m0 + mt * 16 + hi8 + r;
            out[(size_t)m * VOCAB + n0 + r16] = acc[mt][r] + bias;
        }
    }
}

// ---------------- launcher ----------------
extern "C" void kernel_launch(void* const* d_in, const int* in_sizes, int n_in,
                              void* d_out, int out_size, void* d_ws, size_t ws_size,
                              hipStream_t stream) {
    const int*   x     = (const int*)  d_in[0];
    const float* emb   = (const float*)d_in[1];
    const float* W_ih  = (const float*)d_in[2];
    const float* W_hh  = (const float*)d_in[3];
    const float* b_ih  = (const float*)d_in[4];
    const float* b_hh  = (const float*)d_in[5];
    const float* W_aff = (const float*)d_in[6];
    const float* b_aff = (const float*)d_in[7];
    float* out = (float*)d_out;

    // workspace layout
    char* ws = (char*)d_ws;
    float*    W_ihT = (float*)(ws);                                // 128 KB
    float*    W_hhT = (float*)(ws + (128 << 10));                  // 256 KB
    _Float16* wv16  = (_Float16*)(ws + (128 << 10) + (256 << 10)); // 5 MB
    _Float16* hs16  = (_Float16*)(ws + (128 << 10) + (256 << 10)
                                     + (size_t)VOCAB * HID * 2);   // 3.2 MB

    k_transpose_wih<<<(XDIM * HID) / 256, 256, 0, stream>>>(W_ih, W_ihT);
    k_transpose_whh<<<(HID * HID) / 256, 256, 0, stream>>>(W_hh, W_hhT);
    k_convert_waff <<<(VOCAB * HID) / 256, 256, 0, stream>>>(W_aff, wv16);

    k_rnn_recur<<<BATCH, HID, 0, stream>>>(x, emb, W_ihT, W_hhT,
                                           b_ih, b_hh, hs16);

    // 100 m-strips x 125 n-blocks, 5 waves (16N each) per block
    k_out_gemm<<<100 * 125, 160, 0, stream>>>(hs16, wv16, b_aff, out);
}
